// MultiHeadAttention_5360119185803
// MI455X (gfx1250) — matrix operand order repaired and checked by reading, not compiled, not run on hardware
//
#include <hip/hip_runtime.h>
#include <hip/hip_bf16.h>

// ---------------------------------------------------------------------------
// MI455X (gfx1250) fused attention pipeline, bf16 WMMA + async-to-LDS staging.
//   B=4, T=S=2048, E=1024, fp32 in/out.
// ---------------------------------------------------------------------------

typedef __attribute__((ext_vector_type(16))) __bf16 v16bf;
typedef __attribute__((ext_vector_type(8)))  __bf16 v8bf;
typedef __attribute__((ext_vector_type(4)))  __bf16 v4bf;
typedef __attribute__((ext_vector_type(8)))  float  v8f;

#define TILE_M   128
#define TILE_N   128
#define KSTEP    64
#define LDS_PAD  80   // row stride in bf16 elems: 160 bytes (16B & 32B aligned)

// Async copy 16 bytes global -> LDS (gfx1250 GLOBAL_LOAD_ASYNC_TO_LDS_B128,
// tracked by ASYNCcnt). LDS operand is the low 32 bits of the flat address
// (LDS aperture keeps the allocation-relative offset in addr[31:0]; hardware
// adds LDS_BASE).
__device__ __forceinline__ void async_copy_b128(const void* gptr, void* lptr)
{
    unsigned int       loff  = (unsigned int)(unsigned long long)(uintptr_t)lptr;
    unsigned long long gaddr = (unsigned long long)(uintptr_t)gptr;
    asm volatile("global_load_async_to_lds_b128 %0, %1, off"
                 :: "v"(loff), "v"(gaddr)
                 : "memory");
}
__device__ __forceinline__ void wait_async0()
{
    asm volatile("s_wait_asynccnt 0" ::: "memory");
}

// ---------------------------------------------------------------------------
// Generic tiled GEMM:  C = scale * (A x B') [+ bias] [+ resid]
//   F32IN:   A (and B) stored fp32 in global, converted to bf16 into LDS
//            (single-buffered; conversion must go through VGPRs).
//            else A,B already bf16 -> double-buffered async-to-LDS staging.
//   B_TRANS: B stored (N x K) row-major (contraction over inner dim, i.e. W).
//            else B stored (K x N) row-major (i.e. V) -> transposed on store.
//   OUT_F32: C fp32, else bf16.
// M,N multiples of 128; K multiple of 64 (true for all call sites here).
// ---------------------------------------------------------------------------
template <bool F32IN, bool B_TRANS, bool OUT_F32, bool BIAS, bool RESID>
__global__ __launch_bounds__(256)
void gemm_wmma(const void* __restrict__ Ap, long lda, long batchA,
               const void* __restrict__ Bp, long ldb, long batchB,
               void* __restrict__ Cp, long ldc, long batchC,
               const float* __restrict__ bias,
               const float* __restrict__ resid,
               int M, int N, int K, float scale)
{
    (void)M; (void)N;
    constexpr int NBUF = F32IN ? 1 : 2;
    __shared__ __bf16 lA[NBUF][TILE_M][LDS_PAD];
    __shared__ __bf16 lB[NBUF][TILE_N][LDS_PAD];

    const int  tid   = threadIdx.x;        // 0..255
    const int  lane  = tid & 31;
    const int  wave  = tid >> 5;           // 0..7
    const int  waveM = wave >> 1;          // 0..3 -> 32-row strips
    const int  waveN = wave & 1;           // 0..1 -> 64-col strips
    const int  l16   = lane & 15;
    const int  h     = lane >> 4;          // hi half of wave

    const long zb    = blockIdx.z;
    const int  tileM = blockIdx.y * TILE_M;
    const int  tileN = blockIdx.x * TILE_N;

    // ---- stage one 64-deep K chunk into LDS buffer `buf` -------------------
    auto stage = [&](int buf, int kb) {
        // A tile (128 x 64)
        if constexpr (F32IN) {
            const float* Ag = (const float*)Ap + zb * batchA + (long)tileM * lda + kb;
#pragma unroll
            for (int i = 0; i < 8; ++i) {
                int lin = tid + i * 256;          // 0..2047
                int row = lin >> 4;               // 0..127
                int c4  = lin & 15;               // float4 chunk
                const float* src = Ag + (long)row * lda + c4 * 4;
                __builtin_prefetch(src + KSTEP, 0, 0);
                float4 v = *(const float4*)src;
                __bf16* d = &lA[buf][row][c4 * 4];
                d[0] = (__bf16)v.x; d[1] = (__bf16)v.y;
                d[2] = (__bf16)v.z; d[3] = (__bf16)v.w;
            }
        } else {
            const __bf16* Ag = (const __bf16*)Ap + zb * batchA + (long)tileM * lda + kb;
#pragma unroll
            for (int i = 0; i < 4; ++i) {
                int lin = tid + i * 256;          // 0..1023
                int row = lin >> 3;               // 0..127
                int c8  = lin & 7;                // 8-elem (16B) chunk
                async_copy_b128(Ag + (long)row * lda + c8 * 8, &lA[buf][row][c8 * 8]);
            }
        }
        // B tile -> LDS layout [n][k]
        if constexpr (B_TRANS) {
            if constexpr (F32IN) {
                const float* Bg = (const float*)Bp + zb * batchB + (long)tileN * ldb + kb;
#pragma unroll
                for (int i = 0; i < 8; ++i) {
                    int lin = tid + i * 256;
                    int row = lin >> 4;
                    int c4  = lin & 15;
                    const float* src = Bg + (long)row * ldb + c4 * 4;
                    __builtin_prefetch(src + KSTEP, 0, 0);
                    float4 v = *(const float4*)src;
                    __bf16* d = &lB[buf][row][c4 * 4];
                    d[0] = (__bf16)v.x; d[1] = (__bf16)v.y;
                    d[2] = (__bf16)v.z; d[3] = (__bf16)v.w;
                }
            } else {
                const __bf16* Bg = (const __bf16*)Bp + zb * batchB + (long)tileN * ldb + kb;
#pragma unroll
                for (int i = 0; i < 4; ++i) {
                    int lin = tid + i * 256;
                    int row = lin >> 3;
                    int c8  = lin & 7;
                    async_copy_b128(Bg + (long)row * ldb + c8 * 8, &lB[buf][row][c8 * 8]);
                }
            }
        } else {   // (K x N) bf16: transpose during store (VALU path)
            const __bf16* Bg = (const __bf16*)Bp + zb * batchB + (long)kb * ldb + tileN;
#pragma unroll
            for (int i = 0; i < 8; ++i) {
                int lin = tid + i * 256;  // 0..2047
                int k   = lin >> 5;       // 0..63
                int c   = lin & 31;       // 4-elem chunk along n
                const __bf16* src = Bg + (long)k * ldb + c * 4;
                __builtin_prefetch(src + KSTEP * ldb, 0, 0);
                v4bf t = *(const v4bf*)src;
                lB[buf][c * 4 + 0][k] = t[0];
                lB[buf][c * 4 + 1][k] = t[1];
                lB[buf][c * 4 + 2][k] = t[2];
                lB[buf][c * 4 + 3][k] = t[3];
            }
        }
    };

    // ---- WMMA over one staged 64-deep chunk --------------------------------
    v8f acc[2][4];
#pragma unroll
    for (int mt = 0; mt < 2; ++mt)
#pragma unroll
        for (int nt = 0; nt < 4; ++nt)
#pragma unroll
            for (int r = 0; r < 8; ++r) acc[mt][nt][r] = 0.0f;

    auto compute = [&](int buf) {
#pragma unroll
        for (int kk = 0; kk < 2; ++kk) {
            const int k0 = kk * 32;
            v16bf af[2];
#pragma unroll
            for (int mt = 0; mt < 2; ++mt) {
                const __bf16* pa = &lA[buf][waveM * 32 + mt * 16 + l16][k0 + 8 * h];
                v8bf lo = *(const v8bf*)pa;        // K = base .. base+7
                v8bf hi = *(const v8bf*)(pa + 16); // K = base+16 .. base+23
                v16bf a;
#pragma unroll
                for (int i = 0; i < 8; ++i) { a[i] = lo[i]; a[i + 8] = hi[i]; }
                af[mt] = a;
            }
#pragma unroll
            for (int nt = 0; nt < 4; ++nt) {
                const __bf16* pb = &lB[buf][waveN * 64 + nt * 16 + l16][k0 + 8 * h];
                v8bf blo = *(const v8bf*)pb;
                v8bf bhi = *(const v8bf*)(pb + 16);
                v16bf b;
#pragma unroll
                for (int i = 0; i < 8; ++i) { b[i] = blo[i]; b[i + 8] = bhi[i]; }
#pragma unroll
                for (int mt = 0; mt < 2; ++mt) {
                    acc[mt][nt] = __builtin_amdgcn_wmma_f32_16x16x32_bf16(
                        false, af[mt], false, b, (short)0, acc[mt][nt], false, false);
                }
            }
        }
    };

    // ---- main K loop -------------------------------------------------------
    if constexpr (NBUF == 2) {
        stage(0, 0);                                 // prologue: async chunk 0
        int it = 0;
        for (int kb = 0; kb < K; kb += KSTEP, ++it) {
            const int cur = it & 1;
            wait_async0();                           // chunk `cur` landed (this wave)
            __syncthreads();                         // ... and for all waves
            if (kb + KSTEP < K)
                stage(cur ^ 1, kb + KSTEP);          // overlap next chunk with WMMA
            compute(cur);
            __syncthreads();                         // done reading buf `cur`
        }
    } else {
        for (int kb = 0; kb < K; kb += KSTEP) {
            stage(0, kb);
            __syncthreads();
            compute(0);
            __syncthreads();
        }
    }

    // ---- epilogue ----------------------------------------------------------
    float*  Cf = (float*)Cp  + zb * batchC;
    __bf16* Cb = (__bf16*)Cp + zb * batchC;
#pragma unroll
    for (int mt = 0; mt < 2; ++mt) {
#pragma unroll
        for (int nt = 0; nt < 4; ++nt) {
            const int col = tileN + waveN * 64 + nt * 16 + l16;
#pragma unroll
            for (int r = 0; r < 8; ++r) {
                const long row = tileM + waveM * 32 + mt * 16 + h * 8 + r;
                float v = acc[mt][nt][r] * scale;
                if constexpr (BIAS)  v += bias[col];
                if constexpr (RESID) v += resid[row * ldc + col];
                if constexpr (OUT_F32) Cf[row * ldc + col] = v;
                else                   Cb[row * ldc + col] = (__bf16)v;
            }
        }
    }
}

// ---------------------------------------------------------------------------
// Row softmax over S=2048 fp32 scores; writes bf16 probs IN PLACE at the
// start of each row's fp32 slot (so P row stride = 2*S bf16 elems).
// One 256-thread block per row; the whole row lives in registers.
// ---------------------------------------------------------------------------
__global__ __launch_bounds__(256)
void softmax_rows(float* __restrict__ scores, int ncols)
{
    const long row = blockIdx.x;
    float*  src = scores + row * (long)ncols;
    __bf16* dst = (__bf16*)src;
    const int t = threadIdx.x;

    float v[8];
    float m = -3.0e38f;
#pragma unroll
    for (int i = 0; i < 8; ++i) {
        v[i] = src[t + i * 256];
        m = fmaxf(m, v[i]);
    }
    __shared__ float red[256];
    red[t] = m; __syncthreads();
    for (int s = 128; s > 0; s >>= 1) {
        if (t < s) red[t] = fmaxf(red[t], red[t + s]);
        __syncthreads();
    }
    m = red[0]; __syncthreads();

    float sum = 0.0f;
#pragma unroll
    for (int i = 0; i < 8; ++i) { v[i] = __expf(v[i] - m); sum += v[i]; }
    red[t] = sum; __syncthreads();
    for (int s = 128; s > 0; s >>= 1) {
        if (t < s) red[t] += red[t + s];
        __syncthreads();
    }
    const float inv = 1.0f / red[0];
#pragma unroll
    for (int i = 0; i < 8; ++i) dst[t + i * 256] = (__bf16)(v[i] * inv);
}

// ---------------------------------------------------------------------------
// Row LayerNorm over E=1024, in place on d_out. One block per row.
// ---------------------------------------------------------------------------
__global__ __launch_bounds__(256)
void layernorm_rows(float* __restrict__ out,
                    const float* __restrict__ gamma,
                    const float* __restrict__ beta)
{
    const long row = blockIdx.x;
    float* p = out + row * 1024;
    const int t = threadIdx.x;

    float v[4];
    float s = 0.0f, sq = 0.0f;
#pragma unroll
    for (int i = 0; i < 4; ++i) {
        v[i] = p[t + i * 256];
        s += v[i];
        sq += v[i] * v[i];
    }
    __shared__ float rs[256];
    __shared__ float rq[256];
    rs[t] = s; rq[t] = sq; __syncthreads();
    for (int k = 128; k > 0; k >>= 1) {
        if (t < k) { rs[t] += rs[t + k]; rq[t] += rq[t + k]; }
        __syncthreads();
    }
    const float mu  = rs[0] * (1.0f / 1024.0f);
    const float var = rq[0] * (1.0f / 1024.0f) - mu * mu;
    const float rr  = rsqrtf(var + 1e-6f);
#pragma unroll
    for (int i = 0; i < 4; ++i) {
        const int c = t + i * 256;
        p[c] = (v[i] - mu) * rr * gamma[c] + beta[c];
    }
}

// ---------------------------------------------------------------------------
// Host-side launch
// ---------------------------------------------------------------------------
extern "C" void kernel_launch(void* const* d_in, const int* in_sizes, int n_in,
                              void* d_out, int out_size, void* d_ws, size_t ws_size,
                              hipStream_t stream)
{
    (void)in_sizes; (void)n_in; (void)out_size; (void)ws_size;
    constexpr int  B = 4, T = 2048, S = 2048, E = 1024;
    constexpr long BT = (long)B * T;

    const float* query = (const float*)d_in[0];
    const float* key   = (const float*)d_in[1];
    const float* value = (const float*)d_in[2];
    const float* Wq    = (const float*)d_in[3];
    const float* bq    = (const float*)d_in[4];
    const float* Wk    = (const float*)d_in[5];
    const float* bk    = (const float*)d_in[6];
    const float* Wv    = (const float*)d_in[7];
    const float* bv    = (const float*)d_in[8];
    const float* Wo    = (const float*)d_in[9];
    const float* bo    = (const float*)d_in[10];
    const float* gamma = (const float*)d_in[11];
    const float* beta  = (const float*)d_in[12];
    float* out = (float*)d_out;

    char* ws = (char*)d_ws;
    __bf16* Qb     = (__bf16*)(ws);                               // 16 MiB
    __bf16* Kb     = (__bf16*)(ws + 16777216L);                   // 16 MiB
    __bf16* Vb     = (__bf16*)(ws + 2 * 16777216L);               // 16 MiB
    float*  scores = (float*) (ws + 3 * 16777216L);               // 64 MiB (B,T,S)
    float*  ctx    = (float*) (ws + 3 * 16777216L + 67108864L);   // 32 MiB (B,T,E)

    const dim3 blk(256);
    const float inv_sqrt_d = 0.03125f;   // 1/sqrt(1024)

    // 1) Q/K/V projections: (BT x E) = (BT x E) x W^T + b  -> bf16
    {
        dim3 grd(E / TILE_N, BT / TILE_M, 1);
        gemm_wmma<true, true, false, true, false><<<grd, blk, 0, stream>>>(
            query, E, 0, Wq, E, 0, Qb, E, 0, bq, nullptr, (int)BT, E, E, 1.0f);
        gemm_wmma<true, true, false, true, false><<<grd, blk, 0, stream>>>(
            key,   E, 0, Wk, E, 0, Kb, E, 0, bk, nullptr, (int)BT, E, E, 1.0f);
        gemm_wmma<true, true, false, true, false><<<grd, blk, 0, stream>>>(
            value, E, 0, Wv, E, 0, Vb, E, 0, bv, nullptr, (int)BT, E, E, 1.0f);
    }

    // 2) scores = Q x K^T / sqrt(E)  (batched over B)  -> fp32
    {
        dim3 grd(S / TILE_N, T / TILE_M, B);
        gemm_wmma<false, true, true, false, false><<<grd, blk, 0, stream>>>(
            Qb, E, (long)T * E, Kb, E, (long)S * E,
            scores, S, (long)T * S, nullptr, nullptr, T, S, E, inv_sqrt_d);
    }

    // 3) softmax rows -> bf16 probs in place (row stride 2*S bf16 elems)
    softmax_rows<<<dim3(B * T), blk, 0, stream>>>(scores, S);

    // 4) ctx = P x V  (batched) -> fp32
    {
        dim3 grd(E / TILE_N, T / TILE_M, B);
        gemm_wmma<false, false, true, false, false><<<grd, blk, 0, stream>>>(
            (const __bf16*)scores, 2L * S, (long)T * 2 * S,   // P: bf16 view of score rows
            Vb, E, (long)S * E,
            ctx, E, (long)T * E, nullptr, nullptr, T, E, S, 1.0f);
    }

    // 5) out = ctx x Wo^T + bo + residual -> fp32 into d_out
    {
        dim3 grd(E / TILE_N, BT / TILE_M, 1);
        gemm_wmma<true, true, true, true, true><<<grd, blk, 0, stream>>>(
            ctx, E, 0, Wo, E, 0, out, E, 0, bo, query, (int)BT, E, E, 1.0f);
    }

    // 6) LayerNorm rows in place
    layernorm_rows<<<dim3(B * T), blk, 0, stream>>>(out, gamma, beta);
}
